// CCAModule_6416681140735
// MI455X (gfx1250) — matrix operand order repaired and checked by reading, not compiled, not run on hardware
//
#include <hip/hip_runtime.h>
#include <hip/hip_bf16.h>

typedef __attribute__((ext_vector_type(16))) __bf16 v16bf;
typedef __attribute__((ext_vector_type(8)))  float  v8f;

#define CC 4
#define BB 2
#define DD 128
#define FF 256
#define TT 256
#define HH 4
#define TM 32
#define FT (FF*TT)        /* 65536 */
#define XN_STRIDE 136     /* halfs per row; 272B rows, 16B aligned, conflict-friendly */

union U2 { v16bf v; uint4 u[2]; };

// A fragment (16x32 bf16, rows = samples): lanes<16 hold K kk+0..7 and kk+16..23,
// lanes>=16 hold K kk+8..15 and kk+24..31 (per CDNA5 ISA 7.12.2).
__device__ __forceinline__ v16bf load_a_frag(const __bf16* Abase, int rh, int kk, int lane) {
  int m  = (rh << 4) + (lane & 15);
  int k0 = kk + ((lane >> 4) << 3);
  const __bf16* p = Abase + m * XN_STRIDE + k0;
  U2 r;
  r.u[0] = *(const uint4*)(p);
  r.u[1] = *(const uint4*)(p + 16);
  return r.v;
}

// B fragment (32x16 bf16): lane n -> column jb+tile*16+(n&15); lanes<16 K kk+0..7 and kk+16..23,
// lanes>=16 K kk+8..15 and kk+24..31; W stored row-major [j][k].
__device__ __forceinline__ v16bf load_b_frag(const __bf16* W, int jb, int tile, int kk, int lane) {
  int j  = jb + (tile << 4) + (lane & 15);
  int k0 = kk + ((lane >> 4) << 3);
  const __bf16* p = W + j * DD + k0;
  U2 r;
  r.u[0] = *(const uint4*)(p);
  r.u[1] = *(const uint4*)(p + 16);
  return r.v;
}

// 32x128 += (32x128) x (128x128)^T restricted to this wave's (row-half, 32-col) tile.
__device__ __forceinline__ void gemm_tile(const __bf16* Abase, const __bf16* W,
                                          int rh, int jb, int lane, v8f& acc0, v8f& acc1) {
#pragma unroll
  for (int kk = 0; kk < DD; kk += 32) {
    v16bf a  = load_a_frag(Abase, rh, kk, lane);
    v16bf b0 = load_b_frag(W, jb, 0, kk, lane);
    v16bf b1 = load_b_frag(W, jb, 1, kk, lane);
    acc0 = __builtin_amdgcn_wmma_f32_16x16x32_bf16(false, a, false, b0, (short)0, acc0, false, false);
    acc1 = __builtin_amdgcn_wmma_f32_16x16x32_bf16(false, a, false, b1, (short)0, acc1, false, false);
  }
}

// ---------------------------------------------------------------------------
// Prep: fused effective weights (bf16) + biases.
//   W{q,k,v}_eff[j][k] = g[k] * sum_d Wi[j][d] * W[d][k]
//   b_eff[j] = sum_d Wi[j][d]*(b[d] + sum_k W[d][k]*beta[k]) + bi[j]
// blockIdx.x: 0=q, 1=k, 2=v, 3=out_w bf16 copy. 128 threads.
// ---------------------------------------------------------------------------
__global__ __launch_bounds__(128) void cca_prep(
    const float* ln_q_g, const float* ln_q_b, const float* ln_kv_g, const float* ln_kv_b,
    const float* Wq, const float* bq, const float* Wk, const float* bk,
    const float* Wv, const float* bv, const float* in_w, const float* in_b,
    const float* out_w,
    __bf16* Wq_e, __bf16* Wk_e, __bf16* Wv_e, __bf16* Wo_e, float* bias_e) {
  int which = blockIdx.x, tid = threadIdx.x;
  __shared__ float t1[DD];
  if (which == 3) {
    for (int j = 0; j < DD; ++j) Wo_e[j * DD + tid] = (__bf16)out_w[j * DD + tid];
    return;
  }
  const float* W  = (which == 0) ? Wq : (which == 1) ? Wk : Wv;
  const float* b  = (which == 0) ? bq : (which == 1) ? bk : bv;
  const float* g  = (which == 0) ? ln_q_g : ln_kv_g;
  const float* be = (which == 0) ? ln_q_b : ln_kv_b;
  const float* Wi = in_w + which * DD * DD;
  const float* bi = in_b + which * DD;
  __bf16* We = (which == 0) ? Wq_e : (which == 1) ? Wk_e : Wv_e;

  int k = tid;
  float gk = g[k];
  for (int j = 0; j < DD; ++j) {
    float acc = 0.f;
    for (int d = 0; d < DD; ++d) acc += Wi[j * DD + d] * W[d * DD + k];
    We[j * DD + k] = (__bf16)(acc * gk);
  }
  float a2 = b[tid];
  for (int kk = 0; kk < DD; ++kk) a2 += W[tid * DD + kk] * be[kk];
  t1[tid] = a2;
  __syncthreads();
  float bb = bi[tid];
  for (int d = 0; d < DD; ++d) bb += Wi[tid * DD + d] * t1[d];
  bias_e[which * DD + tid] = bb;
}

// ---------------------------------------------------------------------------
// Main fused kernel: 256 threads (8 wave32) per block, 32 t-rows per block.
// grid = B * F * (T/32) = 4096 blocks.
// ---------------------------------------------------------------------------
#define OFS_XN     0                     /* [4][32][136] bf16 = 34816B (overlaid by outbuf later) */
#define OFS_PS1    34816                 /* [8][32] f32 */
#define OFS_PS2    35840
#define OFS_MEAN   36864                 /* [32] f32 */
#define OFS_RSTD   36992
#define OFS_SCORE  37120                 /* [32][4][4] f32 */
#define OFS_ATTN   39168
#define OFS_CTXBF  41216                 /* [32][136] bf16 = 8704B */
#define LDS_BYTES  49920

__global__ __launch_bounds__(256) void cca_main(
    const float* __restrict__ h_all,
    const __bf16* __restrict__ Wq_e, const __bf16* __restrict__ Wk_e,
    const __bf16* __restrict__ Wv_e, const __bf16* __restrict__ Wo_e,
    const float* __restrict__ bias_e, const float* __restrict__ out_b,
    float* __restrict__ out) {
  __shared__ __align__(16) unsigned char lds[LDS_BYTES];
  __bf16* Xn      = (__bf16*)(lds + OFS_XN);
  float*  outbuf  = (float*)(lds + OFS_XN);      // overlay: [32][132], used after Xn is dead
  float*  ps1     = (float*)(lds + OFS_PS1);
  float*  ps2     = (float*)(lds + OFS_PS2);
  float*  smean   = (float*)(lds + OFS_MEAN);
  float*  srstd   = (float*)(lds + OFS_RSTD);
  float*  scoresL = (float*)(lds + OFS_SCORE);
  float*  attnL   = (float*)(lds + OFS_ATTN);
  __bf16* ctxbf   = (__bf16*)(lds + OFS_CTXBF);

  const int tid = threadIdx.x, lane = tid & 31, wv = tid >> 5;
  const int bx = blockIdx.x;
  const int tch = bx & 7, f = (bx >> 3) & 255, b = bx >> 11;
  const int t0 = tch << 5;

  // ---- Phase 0: load + LayerNorm stats + normalized bf16 activations -------
  for (int c = 0; c < CC; ++c) {
    const float* src = h_all + (size_t)(((c * BB + b) * DD + wv * 16) * FF + f) * TT + t0 + lane;
    float x[16], s1 = 0.f, s2 = 0.f;
#pragma unroll
    for (int i = 0; i < 16; ++i) {
      float v = src[(size_t)i * FT];
      x[i] = v; s1 += v; s2 += v * v;
    }
    ps1[wv * 32 + lane] = s1;
    ps2[wv * 32 + lane] = s2;
    __syncthreads();
    if (tid < 32) {
      float a = 0.f, q = 0.f;
      for (int w2 = 0; w2 < 8; ++w2) { a += ps1[w2 * 32 + tid]; q += ps2[w2 * 32 + tid]; }
      float mean = a * (1.f / 128.f);
      float var  = q * (1.f / 128.f) - mean * mean;
      smean[tid] = mean;
      srstd[tid] = rsqrtf(var + 1e-5f);
    }
    __syncthreads();
    float mean = smean[lane], rs = srstd[lane];
    __bf16* dst = Xn + (c * TM + lane) * XN_STRIDE + wv * 16;
#pragma unroll
    for (int i = 0; i < 16; ++i) dst[i] = (__bf16)((x[i] - mean) * rs);
    __syncthreads();
  }

  const int rh = wv & 1, head = wv >> 1, jb = head * 32;
  const int l15 = lane & 15, hi8 = (lane >> 4) << 3;

  // ---- Phase 1: Q = Xn[0] @ Wq_eff^T (this wave's (row-half, head) tile) ---
  v8f q0 = {}, q1 = {};
  gemm_tile(Xn, Wq_e, rh, jb, lane, q0, q1);
  {
    float bq0 = bias_e[jb + l15], bq1 = bias_e[jb + 16 + l15];
#pragma unroll
    for (int r = 0; r < 8; ++r) { q0[r] += bq0; q1[r] += bq1; }
  }

  // ---- Phase 2: K per candidate + scores (register dot + shfl reduce) ------
  const float scale = 0.17677669529663687f;  // 1/sqrt(32)
  for (int c = 0; c < CC; ++c) {
    v8f k0 = {}, k1 = {};
    gemm_tile(Xn + c * TM * XN_STRIDE, Wk_e, rh, jb, lane, k0, k1);
    float bk0 = bias_e[128 + jb + l15], bk1 = bias_e[128 + jb + 16 + l15];
#pragma unroll
    for (int r = 0; r < 8; ++r) {
      float p = q0[r] * (k0[r] + bk0) + q1[r] * (k1[r] + bk1);
      p += __shfl_xor(p, 1, 32);
      p += __shfl_xor(p, 2, 32);
      p += __shfl_xor(p, 4, 32);
      p += __shfl_xor(p, 8, 32);
      if (l15 == 0) {
        int m = rh * 16 + r + hi8;
        scoresL[(m * HH + head) * CC + c] = p * scale;
      }
    }
  }
  __syncthreads();

  // ---- Phase 3: softmax over C (4) -----------------------------------------
  if (tid < TM * HH) {
    int m = tid >> 2, h = tid & 3;
    const float* s = scoresL + (m * HH + h) * CC;
    float s0 = s[0], s1 = s[1], s2 = s[2], s3 = s[3];
    float mx = fmaxf(fmaxf(s0, s1), fmaxf(s2, s3));
    float e0 = __expf(s0 - mx), e1 = __expf(s1 - mx), e2 = __expf(s2 - mx), e3 = __expf(s3 - mx);
    float inv = 1.f / (e0 + e1 + e2 + e3);
    float* a = attnL + (m * HH + h) * CC;
    a[0] = e0 * inv; a[1] = e1 * inv; a[2] = e2 * inv; a[3] = e3 * inv;
  }
  __syncthreads();

  // ---- Phase 4: V per candidate, ctx accumulates in registers --------------
  v8f cx0 = {}, cx1 = {};
  for (int c = 0; c < CC; ++c) {
    v8f v0 = {}, v1 = {};
    gemm_tile(Xn + c * TM * XN_STRIDE, Wv_e, rh, jb, lane, v0, v1);
    float bv0 = bias_e[256 + jb + l15], bv1 = bias_e[256 + jb + 16 + l15];
#pragma unroll
    for (int r = 0; r < 8; ++r) {
      int m = rh * 16 + r + hi8;
      float at = attnL[(m * HH + head) * CC + c];
      cx0[r] += at * (v0[r] + bv0);
      cx1[r] += at * (v1[r] + bv1);
    }
  }

  // ---- Phase 5: stage ctx (bf16) for the out-projection GEMM ---------------
#pragma unroll
  for (int r = 0; r < 8; ++r) {
    int m = rh * 16 + r + hi8;
    ctxbf[m * XN_STRIDE + jb + l15]      = (__bf16)cx0[r];
    ctxbf[m * XN_STRIDE + jb + 16 + l15] = (__bf16)cx1[r];
  }
  __syncthreads();

  // ---- Phase 6: out = ctx @ out_w^T; stage f32 for coalesced writeback -----
  v8f o0 = {}, o1 = {};
  gemm_tile(ctxbf, Wo_e, rh, jb, lane, o0, o1);
#pragma unroll
  for (int r = 0; r < 8; ++r) {
    int m = rh * 16 + r + hi8;
    outbuf[m * 132 + jb + l15]      = o0[r];
    outbuf[m * 132 + jb + 16 + l15] = o1[r];
  }
  __syncthreads();

  // ---- Phase 7: bias + residual + coalesced store (lane -> t) --------------
  {
    const float* res = h_all + (size_t)((b * DD + wv * 16) * FF + f) * TT + t0 + lane;
    float* dst = out + (size_t)((b * DD + wv * 16) * FF + f) * TT + t0 + lane;
#pragma unroll
    for (int i = 0; i < 16; ++i) {
      int j = wv * 16 + i;
      dst[(size_t)i * FT] = outbuf[lane * 132 + j] + out_b[j] + res[(size_t)i * FT];
    }
  }
}

// ---------------------------------------------------------------------------
extern "C" void kernel_launch(void* const* d_in, const int* in_sizes, int n_in,
                              void* d_out, int out_size, void* d_ws, size_t ws_size,
                              hipStream_t stream) {
  const float* h_all   = (const float*)d_in[0];
  const float* ln_q_g  = (const float*)d_in[1];
  const float* ln_q_b  = (const float*)d_in[2];
  const float* ln_kv_g = (const float*)d_in[3];
  const float* ln_kv_b = (const float*)d_in[4];
  const float* Wq      = (const float*)d_in[5];
  const float* bq      = (const float*)d_in[6];
  const float* Wk      = (const float*)d_in[7];
  const float* bk      = (const float*)d_in[8];
  const float* Wv      = (const float*)d_in[9];
  const float* bv      = (const float*)d_in[10];
  const float* in_w    = (const float*)d_in[11];
  const float* in_b    = (const float*)d_in[12];
  const float* out_w   = (const float*)d_in[13];
  const float* out_b   = (const float*)d_in[14];

  // Workspace layout (needs ~132KB):
  char* ws = (char*)d_ws;
  __bf16* Wq_e  = (__bf16*)(ws + 0);
  __bf16* Wk_e  = (__bf16*)(ws + 32768);
  __bf16* Wv_e  = (__bf16*)(ws + 65536);
  __bf16* Wo_e  = (__bf16*)(ws + 98304);
  float*  bias_e = (float*)(ws + 131072);

  cca_prep<<<4, 128, 0, stream>>>(ln_q_g, ln_q_b, ln_kv_g, ln_kv_b,
                                  Wq, bq, Wk, bk, Wv, bv, in_w, in_b, out_w,
                                  Wq_e, Wk_e, Wv_e, Wo_e, bias_e);

  const int nblocks = BB * FF * (TT / TM);  // 4096
  cca_main<<<nblocks, 256, 0, stream>>>(h_all, Wq_e, Wk_e, Wv_e, Wo_e,
                                        bias_e, out_b, (float*)d_out);
}
